// SparseCrossAttention_18846316495023
// MI455X (gfx1250) — matrix operand order repaired and checked by reading, not compiled, not run on hardware
//
#include <hip/hip_runtime.h>

#define BB 4
#define CC 64
#define NN 4096
#define TOPKK 1024
#define MT 16
#define NWAVES 4

typedef __attribute__((ext_vector_type(16))) __bf16 v16bf;
typedef __attribute__((ext_vector_type(8)))  __bf16 v8bf;
typedef __attribute__((ext_vector_type(8)))  float  v8f;

// ---- WMMA operand builders -------------------------------------------------
// A (16x32 bf16, MxK): lane L<16 -> row L, K = 0..7 & 16..23 ; lane L>=16 -> row L-16, K = 8..15 & 24..31
static __device__ __forceinline__ v16bf load_bf_A(const __bf16* rowptr, int lane) {
  const int kb = (lane >> 4) * 8;
  v8bf lo = *(const v8bf*)(rowptr + kb);
  v8bf hi = *(const v8bf*)(rowptr + kb + 16);
  v16bf r;
#pragma unroll
  for (int i = 0; i < 8; ++i) { r[i] = lo[i]; r[i + 8] = hi[i]; }
  return r;
}

// B (32x16 bf16, KxN): lane L -> col L&15, K = 0..7 & 16..23 (L<16) or 8..15 & 24..31 (L>=16) (column-major storage)
static __device__ __forceinline__ v16bf load_bf_B(const __bf16* colptr, int lane) {
  const __bf16* p = colptr + (lane >> 4) * 8;
  v8bf lo = *(const v8bf*)(p);
  v8bf hi = *(const v8bf*)(p + 16);
  v16bf r;
#pragma unroll
  for (int i = 0; i < 8; ++i) { r[i] = lo[i]; r[i + 8] = hi[i]; }
  return r;
}

// A built from f32 LDS row (explicit float4 loads -> ds_load_b128, then cvt)
static __device__ __forceinline__ v16bf build_A_from_f32(const float* p) {
  const float4 f0 = *(const float4*)(p);
  const float4 f1 = *(const float4*)(p + 4);
  const float4 g0 = *(const float4*)(p + 16);
  const float4 g1 = *(const float4*)(p + 20);
  v16bf r;
  r[0]  = (__bf16)f0.x; r[1]  = (__bf16)f0.y; r[2]  = (__bf16)f0.z; r[3]  = (__bf16)f0.w;
  r[4]  = (__bf16)f1.x; r[5]  = (__bf16)f1.y; r[6]  = (__bf16)f1.z; r[7]  = (__bf16)f1.w;
  r[8]  = (__bf16)g0.x; r[9]  = (__bf16)g0.y; r[10] = (__bf16)g0.z; r[11] = (__bf16)g0.w;
  r[12] = (__bf16)g1.x; r[13] = (__bf16)g1.y; r[14] = (__bf16)g1.z; r[15] = (__bf16)g1.w;
  return r;
}

#define WMMA_BF16(A, Bm, Cacc) \
  __builtin_amdgcn_wmma_f32_16x16x32_bf16(false, (A), false, (Bm), (short)0, (Cacc), false, false)

// ---- Kernel 0a: (B,C,N) f32 -> (B,N,C) bf16 tiled transpose ----------------
__global__ __launch_bounds__(256)
void transpose_cvt_kernel(const float* __restrict__ in, __bf16* __restrict__ outp) {
  __shared__ float t[32][33];
  const int b  = blockIdx.z;
  const int n0 = blockIdx.x * 32;
  const int c0 = blockIdx.y * 32;
  const int tx = threadIdx.x, ty = threadIdx.y;
#pragma unroll
  for (int j = 0; j < 4; ++j)
    t[ty + 8 * j][tx] = in[((size_t)b * CC + c0 + ty + 8 * j) * NN + n0 + tx];
  __syncthreads();
#pragma unroll
  for (int j = 0; j < 4; ++j)
    outp[((size_t)b * NN + n0 + ty + 8 * j) * CC + c0 + tx] = (__bf16)t[tx][ty + 8 * j];
}

// ---- Kernel 0b: weight f32 -> bf16 -----------------------------------------
__global__ void convert_w_kernel(const float* __restrict__ wqkv, const float* __restrict__ wout,
                                 __bf16* __restrict__ wqkv_bf, __bf16* __restrict__ wout_bf) {
  for (int i = threadIdx.x; i < 3 * CC * CC; i += blockDim.x) wqkv_bf[i] = (__bf16)wqkv[i];
  for (int i = threadIdx.x; i < CC * CC; i += blockDim.x) wout_bf[i] = (__bf16)wout[i];
}

// ---- Kernel 1: QKV projections (WMMA) --------------------------------------
__global__ __launch_bounds__(256)
void qkv_proj_kernel(const __bf16* __restrict__ xT, const __bf16* __restrict__ sT,
                     const __bf16* __restrict__ wbf,
                     __bf16* __restrict__ q, __bf16* __restrict__ kT, __bf16* __restrict__ vc) {
  const int lane = threadIdx.x & 31;
  const int wave = (int)((blockIdx.x * blockDim.x + threadIdx.x) >> 5);
  const int ot = wave % 12;
  const int nt = (wave / 12) % (NN / 16);
  const int b  = wave / (12 * (NN / 16));
  const int n0 = nt * 16, o0 = ot * 16;

  const __bf16* in = (ot < 4) ? xT : sT;
  const __bf16* ar = in  + ((size_t)b * NN + n0 + (lane & 15)) * CC;
  const __bf16* bc = wbf + (size_t)(o0 + (lane & 15)) * CC;

  v16bf a0 = load_bf_A(ar, lane),      a1 = load_bf_A(ar + 32, lane);
  v16bf b0 = load_bf_B(bc, lane),      b1 = load_bf_B(bc + 32, lane);
  v8f acc = {};
  acc = WMMA_BF16(a0, b0, acc);
  acc = WMMA_BF16(a1, b1, acc);

  const int mrow = (lane >> 4) * 8;
  const int ocol = o0 + (lane & 15);
  if (ot < 8) {
    __bf16* dst = (ot < 4)
        ? (q  + ((size_t)b * NN + n0 + mrow) * CC + ocol)
        : (kT + ((size_t)b * NN + n0 + mrow) * CC + (ocol - 64));
#pragma unroll
    for (int r = 0; r < 8; ++r) dst[(size_t)r * CC] = (__bf16)acc[r];
  } else {
    v8bf pk;  // consecutive rows are consecutive n in (C,N) layout -> one b128 store
#pragma unroll
    for (int r = 0; r < 8; ++r) pk[r] = (__bf16)acc[r];
    *(v8bf*)(vc + ((size_t)b * CC + (ocol - 128)) * NN + (n0 + mrow)) = pk;
  }
}

// ---- Kernel 2: fused attention ---------------------------------------------
__global__ __launch_bounds__(128, 1)
void attn_fused_kernel(const __bf16* __restrict__ qg, const __bf16* __restrict__ kTg,
                       const __bf16* __restrict__ vg, const __bf16* __restrict__ woutg,
                       const float* __restrict__ bout, const float* __restrict__ gamma,
                       const float* __restrict__ xg, float* __restrict__ outg) {
  __shared__ float    sc[MT][NN];          // 256 KB score strip (320 KB LDS/WGP)
  __shared__ float    rinv[MT];
  __shared__ unsigned hist[NWAVES][256];
  __shared__ __bf16   o2[MT][CC];

  const int tid  = threadIdx.x;
  const int w    = tid >> 5;
  const int lane = tid & 31;
  const int b    = blockIdx.x / (NN / MT);
  const int nt   = blockIdx.x % (NN / MT);
  const int n0   = nt * MT;
  const float scale = 0.125f;  // C^-0.5
  const size_t bbase = (size_t)b * NN;

  // ---------------- Phase A: scores = scale * Q K^T (double-buffered) ------
  const __bf16* qrow = qg + (bbase + n0 + (lane & 15)) * CC;
  v16bf qa0 = load_bf_A(qrow, lane);
  v16bf qa1 = load_bf_A(qrow + 32, lane);
  v16bf kb0, kb1;
  {
    const __bf16* kcol = kTg + (bbase + w * 16 + (lane & 15)) * CC;
    kb0 = load_bf_B(kcol, lane);
    kb1 = load_bf_B(kcol + 32, lane);
  }
#pragma unroll 1
  for (int mt = w; mt < NN / 16; mt += NWAVES) {
    const int mtn = mt + NWAVES;
    v16bf nb0 = {}, nb1 = {};
    if (mtn < NN / 16) {
      const __bf16* kcol = kTg + (bbase + mtn * 16 + (lane & 15)) * CC;
      __builtin_prefetch(kcol + (size_t)NWAVES * 16 * CC, 0, 1);  // global_prefetch ~2 tiles ahead
      nb0 = load_bf_B(kcol, lane);
      nb1 = load_bf_B(kcol + 32, lane);
    }
    v8f acc = {};
    acc = WMMA_BF16(qa0, kb0, acc);
    acc = WMMA_BF16(qa1, kb1, acc);
    const int colb = mt * 16 + (lane & 15);
    const int mrow = (lane >> 4) * 8;
#pragma unroll
    for (int r = 0; r < 8; ++r) sc[mrow + r][colb] = acc[r] * scale;
    kb0 = nb0;
    kb1 = nb1;
  }
  __syncthreads();

  // ---------------- Phase B: exact radix top-k threshold + softmax ----------
  for (int j = 0; j < MT / NWAVES; ++j) {
    const int row = w * (MT / NWAVES) + j;
    float* rowp = sc[row];
    unsigned pref = 0;
    unsigned kneed = TOPKK;
    float mraw = -3.4e38f;
#pragma unroll 1
    for (int pass = 0; pass < 4; ++pass) {
      const int shift = 24 - pass * 8;
      *(uint4*)&hist[w][lane * 8]     = make_uint4(0u, 0u, 0u, 0u);
      *(uint4*)&hist[w][lane * 8 + 4] = make_uint4(0u, 0u, 0u, 0u);
      __builtin_amdgcn_wave_barrier();
#pragma unroll 1
      for (int it = 0; it < NN / 128; ++it) {   // lane-contiguous float4 -> ds_load_b128
        const float4 f = *(const float4*)(rowp + lane * 4 + it * 128);
        const float vals[4] = {f.x, f.y, f.z, f.w};
#pragma unroll
        for (int e = 0; e < 4; ++e) {
          const unsigned u = __float_as_uint(vals[e]);
          const unsigned key = (u & 0x80000000u) ? ~u : (u | 0x80000000u);  // monotone key
          if (pass == 0) mraw = fmaxf(mraw, vals[e]);
          const bool match = (pass == 0) || ((key >> (shift + 8)) == pref);
          if (match) atomicAdd(&hist[w][(key >> shift) & 255u], 1u);
        }
      }
      __builtin_amdgcn_wave_barrier();
      // parallel 256-bin suffix scan: 8 bins/lane + wave suffix-sum
      const uint4 h0 = *(const uint4*)&hist[w][lane * 8];
      const uint4 h1 = *(const uint4*)&hist[w][lane * 8 + 4];
      const unsigned bb[8] = {h0.x, h0.y, h0.z, h0.w, h1.x, h1.y, h1.z, h1.w};
      unsigned g = 0;
#pragma unroll
      for (int i = 0; i < 8; ++i) g += bb[i];
      unsigned incl = g;
#pragma unroll
      for (int off = 1; off < 32; off <<= 1) {
        const unsigned v = __shfl_down(incl, off);
        if (lane + off < 32) incl += v;
      }
      unsigned excl = __shfl_down(incl, 1);   // elements in bins above this lane's group
      if (lane == 31) excl = 0;
      const bool found = (excl < kneed) && (excl + g >= kneed);
      unsigned digit = 0, knew = kneed;
      if (found) {
        unsigned cum = excl;
        bool done = false;
#pragma unroll
        for (int i = 7; i >= 0; --i) {
          const unsigned cnt = bb[i];
          if (!done && (cum + cnt >= kneed)) {
            digit = (unsigned)(lane * 8 + i);
            knew = kneed - cum;
            done = true;
          }
          if (!done) cum += cnt;
        }
      }
      const unsigned long long bal = __ballot((int)found);
      const int src = (bal != 0ull) ? (__ffsll(bal) - 1) : 0;
      digit = (unsigned)__shfl((int)digit, src);
      kneed = (unsigned)__shfl((int)knew, src);
      pref = (pref << 8) | digit;
      __builtin_amdgcn_wave_barrier();
    }
    const unsigned tu = (pref & 0x80000000u) ? (pref ^ 0x80000000u) : ~pref;
    const float thr = __uint_as_float(tu);   // exact 1024th-largest score
#pragma unroll
    for (int off = 16; off > 0; off >>= 1) mraw = fmaxf(mraw, __shfl_xor(mraw, off));
    const float m = fmaxf(mraw, 0.0f);       // masked max: zeros always present
    // fused mask-to-zero + exp + sum (single rw sweep)
    float sloc = 0.0f;
#pragma unroll 1
    for (int it = 0; it < NN / 128; ++it) {
      float* p = rowp + lane * 4 + it * 128;
      const float4 f = *(const float4*)p;
      float4 e;
      e.x = __expf(((f.x >= thr) ? f.x : 0.0f) - m);
      e.y = __expf(((f.y >= thr) ? f.y : 0.0f) - m);
      e.z = __expf(((f.z >= thr) ? f.z : 0.0f) - m);
      e.w = __expf(((f.w >= thr) ? f.w : 0.0f) - m);
      sloc += e.x + e.y + e.z + e.w;
      *(float4*)p = e;
    }
#pragma unroll
    for (int off = 16; off > 0; off >>= 1) sloc += __shfl_xor(sloc, off);
    if (lane == 0) rinv[row] = 1.0f / sloc;
  }
  __syncthreads();

  // ---------------- Phase C: out2[n][c] = softmax(row) . V^T (dbl-buffered) -
  {
    v8f acc = {};
    const int ct = w;
    const float* rowp = sc[lane & 15];
    const __bf16* vcol = vg + ((size_t)b * CC + ct * 16 + (lane & 15)) * NN;
    const int kb = (lane >> 4) * 8;
    v16bf vb = load_bf_B(vcol, lane);
#pragma unroll 1
    for (int ks = 0; ks < NN; ks += 32) {
      v16bf nvb = {};
      if (ks + 32 < NN) nvb = load_bf_B(vcol + ks + 32, lane);
      if ((ks & 255) == 0) __builtin_prefetch(vcol + ks + 512, 0, 1);
      const v16bf aA = build_A_from_f32(rowp + ks + kb);
      acc = WMMA_BF16(aA, vb, acc);
      vb = nvb;
    }
    const int mrow = (lane >> 4) * 8;
    const int cc = ct * 16 + (lane & 15);
#pragma unroll
    for (int r = 0; r < 8; ++r)
      o2[mrow + r][cc] = (__bf16)(acc[r] * rinv[mrow + r]);  // fold 1/sum here
  }
  __syncthreads();

  // ---------------- Phase D: W_out . out2 + b_out, gamma*o + x --------------
  {
    const __bf16* wrow = woutg + (size_t)(w * 16 + (lane & 15)) * CC;
    v16bf a0 = load_bf_A(wrow, lane);
    v16bf a1 = load_bf_A(wrow + 32, lane);
    const __bf16* ocol = &o2[lane & 15][0];
    v16bf b0 = load_bf_B(ocol, lane);
    v16bf b1 = load_bf_B(ocol + 32, lane);
    v8f acc = {};
    acc = WMMA_BF16(a0, b0, acc);
    acc = WMMA_BF16(a1, b1, acc);
    const float g = gamma[0];
    const int n = n0 + (lane & 15);
    const int obase = w * 16 + ((lane >> 4) * 8);
#pragma unroll
    for (int r = 0; r < 8; ++r) {
      const int o = obase + r;
      const size_t idx = ((size_t)b * CC + o) * NN + n;
      outg[idx] = g * (acc[r] + bout[o]) + xg[idx];
    }
  }
}

// ---- Launch ----------------------------------------------------------------
extern "C" void kernel_launch(void* const* d_in, const int* in_sizes, int n_in,
                              void* d_out, int out_size, void* d_ws, size_t ws_size,
                              hipStream_t stream) {
  (void)in_sizes; (void)n_in; (void)out_size; (void)ws_size;
  const float* x     = (const float*)d_in[0];
  const float* skip  = (const float*)d_in[1];
  const float* wqkv  = (const float*)d_in[2];
  const float* wout  = (const float*)d_in[3];
  const float* bout  = (const float*)d_in[4];
  const float* gamma = (const float*)d_in[5];
  float* out = (float*)d_out;

  char* ws = (char*)d_ws;
  size_t off = 0;
  auto alloc = [&](size_t bytes) -> void* {
    void* p = ws + off;
    off += (bytes + 255) & ~(size_t)255;
    return p;
  };
  __bf16* xT      = (__bf16*)alloc((size_t)BB * NN * CC * 2);  // x^T  (B,N,C)
  __bf16* sT      = (__bf16*)alloc((size_t)BB * NN * CC * 2);  // skip^T
  __bf16* qb      = (__bf16*)alloc((size_t)BB * NN * CC * 2);  // Q (B,N,C)
  __bf16* kTb     = (__bf16*)alloc((size_t)BB * NN * CC * 2);  // K^T (B,N,C)
  __bf16* vcb     = (__bf16*)alloc((size_t)BB * CC * NN * 2);  // V (B,C,N)
  __bf16* wqkv_bf = (__bf16*)alloc((size_t)3 * CC * CC * 2);
  __bf16* wout_bf = (__bf16*)alloc((size_t)CC * CC * 2);

  dim3 tb(32, 8);
  dim3 tg(NN / 32, CC / 32, BB);
  transpose_cvt_kernel<<<tg, tb, 0, stream>>>(x, xT);
  transpose_cvt_kernel<<<tg, tb, 0, stream>>>(skip, sT);
  convert_w_kernel<<<1, 256, 0, stream>>>(wqkv, wout, wqkv_bf, wout_bf);
  qkv_proj_kernel<<<(BB * (NN / 16) * 12) / 8, 256, 0, stream>>>(xT, sT, wqkv_bf, qb, kTb, vcb);
  attn_fused_kernel<<<BB * (NN / MT), 128, 0, stream>>>(qb, kTb, vcb, wout_bf, bout, gamma, x, out);
}
